// MemoryNN_27659589386762
// MI455X (gfx1250) — compile-verified, matrix operand order checked by reading, not run on hardware
//
#include <hip/hip_runtime.h>
#include <hip/hip_bf16.h>
#include <math.h>

#define Hh 2048
#define Nn 16384
#define Dd 1024
#define EPSf 1e-8f

typedef __attribute__((ext_vector_type(2)))  float    v2f;
typedef __attribute__((ext_vector_type(8)))  float    v8f;
typedef __attribute__((ext_vector_type(16))) _Float16 v16h;

#if defined(__has_builtin)
#  if __has_builtin(__builtin_amdgcn_wmma_f32_16x16x4_f32)
#    define HAVE_WMMA_F32_K4 1
#  else
#    define HAVE_WMMA_F32_K4 0
#  endif
#else
#  define HAVE_WMMA_F32_K4 0
#endif

__device__ __forceinline__ float wave_sum(float v) {
    v += __shfl_xor(v, 1, 32);
    v += __shfl_xor(v, 2, 32);
    v += __shfl_xor(v, 4, 32);
    v += __shfl_xor(v, 8, 32);
    v += __shfl_xor(v, 16, 32);
    return v;
}

// ---------------- K1/K3: wave-per-row GEMV (float4 streams, wave32 reduce) ----
__global__ void gemv_k(const float* __restrict__ Wm, const float* __restrict__ x,
                       const float* __restrict__ b, float* __restrict__ out,
                       int rows, int cols4, int sigmoid_mode) {
    int wid  = (blockIdx.x * blockDim.x + threadIdx.x) >> 5;
    int lane = threadIdx.x & 31;
    if (wid >= rows) return;
    const float4* Wr = (const float4*)(Wm + (size_t)wid * (size_t)(cols4 * 4));
    const float4* xr = (const float4*)x;
    float acc = 0.f;
    for (int t = lane; t < cols4; t += 32) {
        float4 w = Wr[t];
        float4 xv = xr[t];
        acc = fmaf(w.x, xv.x, acc);
        acc = fmaf(w.y, xv.y, acc);
        acc = fmaf(w.z, xv.z, acc);
        acc = fmaf(w.w, xv.w, acc);
    }
    acc = wave_sum(acc);
    if (lane == 0) {
        float r = acc + b[wid];
        out[wid] = sigmoid_mode ? (1.f / (1.f + expf(-r))) : r;
    }
}

// ---------------- K2: scalars: beta, gamma, softmax(shift), ||k|| ------------
__global__ void scalars_k(const float* __restrict__ dim_out, float* __restrict__ sc) {
    __shared__ float red[256];
    int t = threadIdx.x;
    float ss = 0.f;
    for (int i = t; i < Dd; i += 256) { float kv = dim_out[5 + i]; ss = fmaf(kv, kv, ss); }
    red[t] = ss; __syncthreads();
    for (int s = 128; s > 0; s >>= 1) { if (t < s) red[t] += red[t + s]; __syncthreads(); }
    if (t == 0) {
        float a0 = dim_out[2], a1 = dim_out[3], a2 = dim_out[4];
        float mx = fmaxf(a0, fmaxf(a1, a2));
        float e0 = expf(a0 - mx), e1 = expf(a1 - mx), e2 = expf(a2 - mx);
        float s = e0 + e1 + e2;
        sc[0] = dim_out[0];          // beta
        sc[1] = dim_out[1];          // gamma
        sc[2] = e0 / s; sc[3] = e1 / s; sc[4] = e2 / s;
        sc[5] = sqrtf(red[0]);       // ||k||
    }
}

// ---------------- K4: sim[i] = <mem_i, k> / (||mem_i||*||k|| + eps) ----------
__global__ void sim_k(const float* __restrict__ mem, const float* __restrict__ dim_out,
                      const float* __restrict__ sc, float* __restrict__ sim) {
    int wid  = (blockIdx.x * blockDim.x + threadIdx.x) >> 5;
    int lane = threadIdx.x & 31;
    if (wid >= Nn) return;
    const float4* Mr = (const float4*)(mem + (size_t)wid * Dd);
    const float*  kp = dim_out + 5;          // 4B aligned only -> scalar loads (cached)
    float dot = 0.f, ssq = 0.f;
    for (int t = lane; t < Dd / 4; t += 32) {
        float4 m = Mr[t];
        int b = t * 4;
        dot = fmaf(m.x, kp[b + 0], dot);
        dot = fmaf(m.y, kp[b + 1], dot);
        dot = fmaf(m.z, kp[b + 2], dot);
        dot = fmaf(m.w, kp[b + 3], dot);
        ssq = fmaf(m.x, m.x, ssq);
        ssq = fmaf(m.y, m.y, ssq);
        ssq = fmaf(m.z, m.z, ssq);
        ssq = fmaf(m.w, m.w, ssq);
    }
    dot = wave_sum(dot);
    ssq = wave_sum(ssq);
    if (lane == 0) sim[wid] = dot / (sqrtf(ssq) * sc[5] + EPSf);
}

// ---------------- K5: softmax -> gate-mix -> shift-conv -> pow -> weight -----
__global__ void __launch_bounds__(1024)
weight_k(const float* __restrict__ sim, const float* __restrict__ gate,
         const float* __restrict__ lastw, const float* __restrict__ sc,
         float* __restrict__ wg, float* __restrict__ weight_out) {
    __shared__ float red[1024];
    int t = threadIdx.x;
    float beta = sc[0], gamma = sc[1], s0 = sc[2], s1 = sc[3], s2 = sc[4];
    float ls[16];
    float lmax = -3.402823466e38f;
    #pragma unroll
    for (int p = 0; p < 16; ++p) {
        float v = beta * sim[t + p * 1024];
        ls[p] = v; lmax = fmaxf(lmax, v);
    }
    red[t] = lmax; __syncthreads();
    for (int s = 512; s > 0; s >>= 1) { if (t < s) red[t] = fmaxf(red[t], red[t + s]); __syncthreads(); }
    float m = red[0]; __syncthreads();

    float lsum = 0.f;
    #pragma unroll
    for (int p = 0; p < 16; ++p) { float e = expf(ls[p] - m); ls[p] = e; lsum += e; }
    red[t] = lsum; __syncthreads();
    for (int s = 512; s > 0; s >>= 1) { if (t < s) red[t] += red[t + s]; __syncthreads(); }
    float invS = 1.f / red[0]; __syncthreads();

    #pragma unroll
    for (int p = 0; p < 16; ++p) {
        int i = t + p * 1024;
        float wc = ls[p] * invS;
        float g  = gate[i];
        wg[i] = fmaf(g, wc, (1.f - g) * lastw[i]);
    }
    __threadfence_block();
    __syncthreads();                 // wg fully visible block-wide (single block)

    float lps = 0.f;
    #pragma unroll
    for (int p = 0; p < 16; ++p) {
        int i  = t + p * 1024;
        int im = (i == 0)      ? Nn - 1 : i - 1;
        int ip = (i == Nn - 1) ? 0      : i + 1;
        float wt = s0 * wg[im] + s1 * wg[i] + s2 * wg[ip];
        float wp = powf(wt, gamma);
        ls[p] = wp; lps += wp;
    }
    red[t] = lps; __syncthreads();
    for (int s = 512; s > 0; s >>= 1) { if (t < s) red[t] += red[t + s]; __syncthreads(); }
    float invT = 1.f / red[0];
    #pragma unroll
    for (int p = 0; p < 16; ++p) weight_out[t + p * 1024] = ls[p] * invT;
}

// ---------------- K6: fused new_memory + memory_read via WMMA outer products -
// 1024 waves. Wave (slot, jc): columns [16*jc,16*jc+16), rows [1024*slot, +1024).
// Per 16x16 tile: C_e = w ⊗ erase, C_a = w ⊗ add via V_WMMA_F32_16X16X4_F32
// (A col0 = weight slice, B row0 = erase/add slice). C/D layout == tile layout.
__global__ void __launch_bounds__(256)
update_k(const float* __restrict__ mem, const float* __restrict__ dim_out,
         const float* __restrict__ weight, float* __restrict__ newmem,
         float* __restrict__ read_part) {
    const float* erase = dim_out + 5 + Dd;
    const float* addv  = dim_out + 5 + 2 * Dd;
    int lane = threadIdx.x & 31;
    int wv   = threadIdx.x >> 5;
    int W    = blockIdx.x * 8 + wv;       // 0..1023
    int jc   = W & 63;                    // 64 column chunks of 16
    int slot = W >> 6;                    // 16 row slots of 64 tiles
    int col  = jc * 16 + (lane & 15);
    bool lo  = lane < 16;
    float ecol = lo ? erase[col] : 0.f;   // B[0][n]; B[1..3][n]=0
    float acol = lo ? addv[col]  : 0.f;
    int rsub = (lane >> 4) << 3;          // rows 0..7 vs 8..15 of the tile

#if HAVE_WMMA_F32_K4
    v2f be; be[0] = ecol; be[1] = 0.f;
    v2f ba; ba[0] = acol; ba[1] = 0.f;
#else
    v16h bhe = {}; bhe[0] = (_Float16)ecol;
    v16h bha = {}; bha[0] = (_Float16)acol;
#endif

    float racc = 0.f;
    for (int it = slot * 64; it < slot * 64 + 64; ++it) {
        int i0 = it << 4;
        float wlane = lo ? weight[i0 + lane] : 0.f;   // A[m][0]; A[m][1..3]=0
        v8f ce, ca;
        v8f cz = {};
#if HAVE_WMMA_F32_K4
        v2f aw; aw[0] = wlane; aw[1] = 0.f;
        ce = __builtin_amdgcn_wmma_f32_16x16x4_f32(false, aw, false, be,
                                                   (short)0, cz, false, false);
        ca = __builtin_amdgcn_wmma_f32_16x16x4_f32(false, aw, false, ba,
                                                   (short)0, cz, false, false);
#else
        v16h ah = {}; ah[0] = (_Float16)wlane;
        ce = __builtin_amdgcn_wmma_f32_16x16x32_f16(false, ah, false, bhe,
                                                    (short)0, cz, false, false);
        ca = __builtin_amdgcn_wmma_f32_16x16x32_f16(false, ah, false, bha,
                                                    (short)0, cz, false, false);
#endif
        int rbase = i0 + rsub;
        const float* mrow = mem    + (size_t)rbase * Dd + col;
        float*       orow = newmem + (size_t)rbase * Dd + col;
        float r = 0.f;
        #pragma unroll
        for (int v = 0; v < 8; ++v) {
            float mval = mrow[(size_t)v * Dd];          // L2-resident after sim pass
            float wrow = weight[rbase + v];
            r = fmaf(wrow, mval, r);                    // weight @ memory partial
            orow[(size_t)v * Dd] = fmaf(mval, -ce[v], mval) + ca[v];  // m*(1-we)+wa
        }
        racc += r;
    }
    racc += __shfl_xor(racc, 16, 32);     // combine rows 0-7 / 8-15 of same column
    if (lo) read_part[slot * Dd + col] = racc;   // race-free: unique (slot,col)
}

// ---------------- K7: deterministic 16-way reduce -> memory_read -------------
__global__ void read_reduce_k(const float* __restrict__ read_part, float* __restrict__ mread) {
    int j = blockIdx.x * blockDim.x + threadIdx.x;
    if (j < Dd) {
        float s = 0.f;
        #pragma unroll
        for (int p = 0; p < 16; ++p) s += read_part[p * Dd + j];
        mread[j] = s;
    }
}

extern "C" void kernel_launch(void* const* d_in, const int* in_sizes, int n_in,
                              void* d_out, int out_size, void* d_ws, size_t ws_size,
                              hipStream_t stream) {
    const float* hidden = (const float*)d_in[0];
    const float* lastw  = (const float*)d_in[1];
    const float* mem    = (const float*)d_in[2];
    const float* W_dim  = (const float*)d_in[3];
    const float* b_dim  = (const float*)d_in[4];
    const float* W_gate = (const float*)d_in[5];
    const float* b_gate = (const float*)d_in[6];

    float* out    = (float*)d_out;
    float* mread  = out;                 // [D]
    float* weight = out + Dd;            // [N]
    float* newmem = out + Dd + Nn;       // [N*D]

    float* ws      = (float*)d_ws;
    float* dim_out = ws;                 // 3077 (pad to 3104)
    float* sc      = ws + 3104;          // 8 scalars
    float* gate    = ws + 3112;          // N
    float* sim     = gate + Nn;          // N
    float* wg      = sim + Nn;           // N
    float* rpart   = wg + Nn;            // 16*D

    const int DIMROWS = 5 + 3 * Dd;      // 3077

    // K1: dim_out = W_dim @ hidden + b_dim   (25 MB stream)
    {
        int blocks = (DIMROWS * 32 + 255) / 256;
        gemv_k<<<blocks, 256, 0, stream>>>(W_dim, hidden, b_dim, dim_out, DIMROWS, Hh / 4, 0);
    }
    // K2: scalars
    scalars_k<<<1, 256, 0, stream>>>(dim_out, sc);
    // K3: gate = sigmoid(W_gate @ hidden + b_gate)   (134 MB stream, dominant)
    gemv_k<<<(Nn * 32) / 256, 256, 0, stream>>>(W_gate, hidden, b_gate, gate, Nn, Hh / 4, 1);
    // K4: sim (64 MB stream; leaves `memory` resident in 192 MB L2)
    sim_k<<<(Nn * 32) / 256, 256, 0, stream>>>(mem, dim_out, sc, sim);
    // K5: weight (single workgroup, 16K elems)
    weight_k<<<1, 1024, 0, stream>>>(sim, gate, lastw, sc, wg, weight);
    // K6: new_memory + read partials (WMMA outer products; memory re-read hits L2)
    update_k<<<128, 256, 0, stream>>>(mem, dim_out, weight, newmem, rpart);
    // K7: memory_read
    read_reduce_k<<<4, 256, 0, stream>>>(rpart, mread);
}
